// DecoderRNN_86517821213807
// MI455X (gfx1250) — matrix operand order repaired and checked by reading, not compiled, not run on hardware
//
#include <hip/hip_runtime.h>
#include <hip/hip_bf16.h>
#include <math.h>

// Problem constants (match reference setup_inputs)
#define HB   512     // hidden
#define EB   512     // embed
#define BB   128     // batch
#define SS   400     // src len
#define VV   50000   // vocab
#define EXTV 50100   // ext vocab

typedef __attribute__((ext_vector_type(16))) __bf16 v16bf;
typedef __attribute__((ext_vector_type(8)))  float  v8f;

__device__ __forceinline__ float sigf(float x) { return 1.0f / (1.0f + expf(-x)); }

// ---- WMMA fragment loaders: convert f32 -> bf16 in registers ----
// A (16x32 bf16): lane L holds row m=L%16; element j: K = 16*half + j
__device__ __forceinline__ v16bf load_a_frag(const float* __restrict__ aRow, int kk, int half) {
    const float* p = aRow + kk + 16 * half;
    float4 x0 = *(const float4*)(p);
    float4 x1 = *(const float4*)(p + 4);
    float4 y0 = *(const float4*)(p + 8);
    float4 y1 = *(const float4*)(p + 12);
    v16bf a;
    a[0]  = (__bf16)x0.x; a[1]  = (__bf16)x0.y; a[2]  = (__bf16)x0.z; a[3]  = (__bf16)x0.w;
    a[4]  = (__bf16)x1.x; a[5]  = (__bf16)x1.y; a[6]  = (__bf16)x1.z; a[7]  = (__bf16)x1.w;
    a[8]  = (__bf16)y0.x; a[9]  = (__bf16)y0.y; a[10] = (__bf16)y0.z; a[11] = (__bf16)y0.w;
    a[12] = (__bf16)y1.x; a[13] = (__bf16)y1.y; a[14] = (__bf16)y1.z; a[15] = (__bf16)y1.w;
    return a;
}
// B (32x16 bf16), source is column-major (per-column contiguous K):
// lane L holds col n=L%16; element j: K = 16*half + j
__device__ __forceinline__ v16bf load_b_frag(const float* __restrict__ bCol, int kk, int half) {
    const float* p = bCol + kk + 16 * half;
    float4 x0 = *(const float4*)(p);
    float4 x1 = *(const float4*)(p + 4);
    float4 x2 = *(const float4*)(p + 8);
    float4 x3 = *(const float4*)(p + 12);
    v16bf b;
    b[0]  = (__bf16)x0.x; b[1]  = (__bf16)x0.y; b[2]  = (__bf16)x0.z; b[3]  = (__bf16)x0.w;
    b[4]  = (__bf16)x1.x; b[5]  = (__bf16)x1.y; b[6]  = (__bf16)x1.z; b[7]  = (__bf16)x1.w;
    b[8]  = (__bf16)x2.x; b[9]  = (__bf16)x2.y; b[10] = (__bf16)x2.z; b[11] = (__bf16)x2.w;
    b[12] = (__bf16)x3.x; b[13] = (__bf16)x3.y; b[14] = (__bf16)x3.z; b[15] = (__bf16)x3.w;
    return b;
}

// ---- Generic GEMM: C(MxN) = A(MxK, f32 row-major) * Bcol(K per column contiguous) [+bias][+=]
// grid = (N/16, M/16), block = 32 (one wave per 16x16 tile). Used for the small GEMMs.
__global__ void gemm_bf16_wmma(const float* __restrict__ A, int lda,
                               const float* __restrict__ B, int ldb,
                               float* __restrict__ C, int ldc,
                               int K, const float* __restrict__ bias, int accum) {
    const int lane = threadIdx.x;
    const int m = lane & 15, half = lane >> 4;
    const int row0 = blockIdx.y << 4;
    const int col0 = blockIdx.x << 4;
    const float* aRow = A + (size_t)(row0 + m) * (size_t)lda;
    const float* bCol = B + (size_t)(col0 + m) * (size_t)ldb;
    v8f acc = {0.f, 0.f, 0.f, 0.f, 0.f, 0.f, 0.f, 0.f};
    for (int kk = 0; kk < K; kk += 32) {
        v16bf a = load_a_frag(aRow, kk, half);
        v16bf b = load_b_frag(bCol, kk, half);
        acc = __builtin_amdgcn_wmma_f32_16x16x32_bf16(false, a, false, b, (short)0, acc, false, false);
    }
    const int col = col0 + m;
    const float bv = bias ? bias[col] : 0.0f;
#pragma unroll
    for (int i = 0; i < 8; ++i) {
        int r = row0 + i + (half << 3);
        size_t o = (size_t)r * (size_t)ldc + col;
        float v = acc[i] + bv;
        if (accum) C[o] += v; else C[o] = v;
    }
}

// ---- Tall-B GEMM for M=128: one wave computes ALL 8 row-tiles for its 16 columns.
// Each B fragment (the huge streamed operand, e.g. out_W) is loaded exactly ONCE from
// memory per wave -> out_W streams through HBM exactly once (it exceeds L2).
// grid = (N/16), block = 32.
__global__ void gemm_m128_wmma(const float* __restrict__ A, int lda,
                               const float* __restrict__ B, int ldb,
                               float* __restrict__ C, int ldc, int K) {
    const int lane = threadIdx.x;
    const int m = lane & 15, half = lane >> 4;
    const int col0 = blockIdx.x << 4;
    const float* bCol = B + (size_t)(col0 + m) * (size_t)ldb;
    v8f acc[8];
#pragma unroll
    for (int t = 0; t < 8; ++t) acc[t] = (v8f){0.f, 0.f, 0.f, 0.f, 0.f, 0.f, 0.f, 0.f};
    for (int kk = 0; kk < K; kk += 32) {
        v16bf b = load_b_frag(bCol, kk, half);
#pragma unroll
        for (int t = 0; t < 8; ++t) {
            v16bf a = load_a_frag(A + (size_t)((t << 4) + m) * (size_t)lda, kk, half);
            acc[t] = __builtin_amdgcn_wmma_f32_16x16x32_bf16(false, a, false, b, (short)0, acc[t], false, false);
        }
    }
    const int col = col0 + m;
#pragma unroll
    for (int t = 0; t < 8; ++t) {
#pragma unroll
        for (int i = 0; i < 8; ++i) {
            int r = (t << 4) + i + (half << 3);
            C[(size_t)r * (size_t)ldc + col] = acc[t][i];
        }
    }
}

// ---- small prep kernels ----
__global__ void concat_kernel(const float* __restrict__ a, const float* __restrict__ b,
                              float* __restrict__ out, int n1, int n2, int rows) {
    int i = blockIdx.x * blockDim.x + threadIdx.x;
    int w = n1 + n2;
    if (i >= rows * w) return;
    int r = i / w, c = i % w;
    out[i] = (c < n1) ? a[(size_t)r * n1 + c] : b[(size_t)r * n2 + (c - n1)];
}

__global__ void transpose_kernel(const float* __restrict__ in, float* __restrict__ out, int R, int C) {
    int i = blockIdx.x * blockDim.x + threadIdx.x;
    if (i >= R * C) return;
    int r = i / C, c = i % C;
    out[(size_t)c * R + r] = in[i];
}

// ---- LSTM pointwise: gates(B,4H) -> h,c; hidden=[h|c] ----
__global__ void lstm_kernel(const float* __restrict__ gates, const float* __restrict__ c_prev,
                            float* __restrict__ h_out, float* __restrict__ c_out,
                            float* __restrict__ hidden) {
    int b = blockIdx.x, j = threadIdx.x;
    const float* g = gates + (size_t)b * 4 * HB;
    float ig = sigf(g[j]);
    float fg = sigf(g[HB + j]);
    float gg = tanhf(g[2 * HB + j]);
    float og = sigf(g[3 * HB + j]);
    float c = fg * c_prev[(size_t)b * HB + j] + ig * gg;
    float h = og * tanhf(c);
    h_out[(size_t)b * HB + j] = h;
    c_out[(size_t)b * HB + j] = c;
    hidden[(size_t)b * 2 * HB + j] = h;
    hidden[(size_t)b * 2 * HB + HB + j] = c;
}

// ---- Fused attention energy: rows r=(b*S+s); per 16-row block, 8 waves x 4 N-tiles,
// energy = tanh(mem@W2T + hidW[b] + cov[r]*coverW), attention[r] = sum_n energy*W3[n]
__global__ void attn_energy_kernel(const float* __restrict__ enc,    // (S,B,H) f32
                                   const float* __restrict__ W2T,    // col-major B (H x H)
                                   const float* __restrict__ hidW,   // (B,H)
                                   const float* __restrict__ cov,    // (B*S) flat
                                   const float* __restrict__ coverW, // (H)
                                   const float* __restrict__ W3,     // (H)
                                   float* __restrict__ attention) {  // (B*S)
    __shared__ float rowAcc[16];
    const int tid = threadIdx.x;
    const int lane = tid & 31;
    const int wid = tid >> 5;          // 0..7
    const int m = lane & 15, half = lane >> 4;
    const int row0 = blockIdx.x << 4;

    if (tid < 16) rowAcc[tid] = 0.0f;
    __syncthreads();

    // A row for this lane (fixed across tiles)
    const int rA = row0 + m;
    const int bA = rA / SS;
    const int sA = rA - bA * SS;
    const float* aRow = enc + ((size_t)sA * BB + bA) * HB;

    int rC[8], bC[8];
#pragma unroll
    for (int i = 0; i < 8; ++i) { rC[i] = row0 + i + (half << 3); bC[i] = rC[i] / SS; }

    float rs[8];
#pragma unroll
    for (int i = 0; i < 8; ++i) rs[i] = 0.0f;

    for (int t = 0; t < 4; ++t) {
        const int n = ((wid << 2) + t) * 16 + m;
        const float* bCol = W2T + (size_t)n * HB;
        v8f acc = {0.f, 0.f, 0.f, 0.f, 0.f, 0.f, 0.f, 0.f};
        for (int kk = 0; kk < HB; kk += 32) {
            v16bf a = load_a_frag(aRow, kk, half);
            v16bf b = load_b_frag(bCol, kk, half);
            acc = __builtin_amdgcn_wmma_f32_16x16x32_bf16(false, a, false, b, (short)0, acc, false, false);
        }
        const float w3 = W3[n];
        const float cw = coverW[n];
#pragma unroll
        for (int i = 0; i < 8; ++i) {
            float e = acc[i] + hidW[(size_t)bC[i] * HB + n] + cov[rC[i]] * cw;
            rs[i] += tanhf(e) * w3;
        }
    }
#pragma unroll
    for (int i = 0; i < 8; ++i) atomicAdd(&rowAcc[i + (half << 3)], rs[i]);
    __syncthreads();
    if (tid < 16) attention[row0 + tid] = rowAcc[tid];
}

// ---- masked softmax over S + context = sum_s p_s * mem[b,s,:]; build combined=[h|ctx] ----
__global__ void softmax_ctx_kernel(const float* __restrict__ attention,
                                   const float* __restrict__ mask,
                                   const float* __restrict__ enc,     // (S,B,H)
                                   const float* __restrict__ h_new,   // (B,H)
                                   float* __restrict__ enc_attn_out,  // (B,S)
                                   float* __restrict__ enc_ctx,       // (B,H)
                                   float* __restrict__ combined) {    // (B,2H)
    __shared__ float sp[SS];
    __shared__ float red[512];
    const int b = blockIdx.x, tid = threadIdx.x;
    float a = -3.0e38f;
    if (tid < SS) {
        float mv = mask[(size_t)b * SS + tid];
        a = mv * attention[(size_t)b * SS + tid] - (1.0f - mv) * 1.0e20f;
    }
    red[tid] = a;
    __syncthreads();
    for (int s2 = 256; s2 > 0; s2 >>= 1) { if (tid < s2) red[tid] = fmaxf(red[tid], red[tid + s2]); __syncthreads(); }
    float mx = red[0];
    __syncthreads();
    float e = 0.0f;
    if (tid < SS) e = expf(a - mx);
    red[tid] = e;
    __syncthreads();
    for (int s2 = 256; s2 > 0; s2 >>= 1) { if (tid < s2) red[tid] += red[tid + s2]; __syncthreads(); }
    float inv = 1.0f / red[0];
    __syncthreads();
    if (tid < SS) {
        float p = e * inv;
        sp[tid] = p;
        enc_attn_out[(size_t)b * SS + tid] = p;
    }
    __syncthreads();
    float ctx = 0.0f;
    for (int s = 0; s < SS; ++s) ctx += sp[s] * enc[((size_t)s * BB + b) * HB + tid];
    enc_ctx[(size_t)b * HB + tid] = ctx;
    combined[(size_t)b * 2 * HB + HB + tid] = ctx;
    combined[(size_t)b * 2 * HB + tid] = h_new[(size_t)b * HB + tid];
}

// ---- pointer prob: sigmoid([h|ctx|emb|h] . ptrW + ptrb) ----
__global__ void ptr_kernel(const float* __restrict__ h_new, const float* __restrict__ enc_ctx,
                           const float* __restrict__ emb, const float* __restrict__ ptrW,
                           const float* __restrict__ ptrb, float* __restrict__ prob_ptr) {
    __shared__ float red[256];
    const int b = blockIdx.x, tid = threadIdx.x;
    float s = 0.0f;
    for (int j = tid; j < 2 * HB + EB + HB; j += 256) {
        float v;
        if (j < HB)               v = h_new[(size_t)b * HB + j];
        else if (j < 2 * HB)      v = enc_ctx[(size_t)b * HB + (j - HB)];
        else if (j < 2 * HB + EB) v = emb[(size_t)b * EB + (j - 2 * HB)];
        else                      v = h_new[(size_t)b * HB + (j - 2 * HB - EB)];
        s += v * ptrW[j];
    }
    red[tid] = s;
    __syncthreads();
    for (int s2 = 128; s2 > 0; s2 >>= 1) { if (tid < s2) red[tid] += red[tid + s2]; __syncthreads(); }
    if (tid == 0) prob_ptr[b] = sigf(red[0] + ptrb[0]);
}

// ---- vocab softmax scaled by (1 - p_ptr); zero ext tail ----
__global__ void vocab_kernel(const float* __restrict__ logits, const float* __restrict__ prob_ptr,
                             float* __restrict__ out_probs) {
    __shared__ float red[512];
    const int b = blockIdx.x, tid = threadIdx.x;
    const float* lg = logits + (size_t)b * VV;
    float mx = -3.0e38f;
    for (int v = tid; v < VV; v += 512) mx = fmaxf(mx, lg[v]);
    red[tid] = mx;
    __syncthreads();
    for (int s2 = 256; s2 > 0; s2 >>= 1) { if (tid < s2) red[tid] = fmaxf(red[tid], red[tid + s2]); __syncthreads(); }
    mx = red[0];
    __syncthreads();
    float sum = 0.0f;
    for (int v = tid; v < VV; v += 512) sum += expf(lg[v] - mx);
    red[tid] = sum;
    __syncthreads();
    for (int s2 = 256; s2 > 0; s2 >>= 1) { if (tid < s2) red[tid] += red[tid + s2]; __syncthreads(); }
    float pg = (1.0f - prob_ptr[b]) / red[0];
    float* op = out_probs + (size_t)b * EXTV;
    for (int v = tid; v < VV; v += 512) op[v] = pg * expf(lg[v] - mx);
    for (int v = VV + tid; v < EXTV; v += 512) op[v] = 0.0f;
}

__global__ void scatter_kernel(const int* __restrict__ idx, const float* __restrict__ enc_attn,
                               const float* __restrict__ prob_ptr, float* __restrict__ out_probs) {
    int i = blockIdx.x * blockDim.x + threadIdx.x;
    if (i >= BB * SS) return;
    int b = i / SS;
    atomicAdd(&out_probs[(size_t)b * EXTV + idx[i]], prob_ptr[b] * enc_attn[i]);
}

__global__ void log_kernel(float* __restrict__ out_probs, int n) {
    int i = blockIdx.x * blockDim.x + threadIdx.x;
    if (i < n) out_probs[i] = logf(out_probs[i] + 1e-31f);
}

extern "C" void kernel_launch(void* const* d_in, const int* in_sizes, int n_in,
                              void* d_out, int out_size, void* d_ws, size_t ws_size,
                              hipStream_t stream) {
    (void)in_sizes; (void)n_in; (void)out_size; (void)ws_size;
    const float* embedded = (const float*)d_in[0];   // (B,E)
    const float* h0       = (const float*)d_in[1];   // (1,B,H)
    const float* c0       = (const float*)d_in[2];   // (1,B,H)
    const float* enc      = (const float*)d_in[3];   // (S,B,H)
    const float* cov      = (const float*)d_in[4];   // (B,S)
    const float* mask     = (const float*)d_in[5];   // (B,S)
    const int*   widx     = (const int*)d_in[6];     // (B,S)
    const float* prevctx  = (const float*)d_in[7];   // (B,H)
    const float* fc_dec_W = (const float*)d_in[9];   // (E, H+E) row-major == colB
    const float* fc_dec_b = (const float*)d_in[10];
    const float* W_ih     = (const float*)d_in[11];  // (4H, E) row-major == colB
    const float* W_hh     = (const float*)d_in[12];  // (4H, H) row-major == colB
    const float* b_ih     = (const float*)d_in[13];
    const float* b_hh     = (const float*)d_in[14];
    const float* attn_W   = (const float*)d_in[15];  // (2H, H)  -> transpose
    const float* attn_W2  = (const float*)d_in[16];  // (H, H)   -> transpose
    const float* attn_W3  = (const float*)d_in[17];  // (H,)
    const float* coverW   = (const float*)d_in[18];  // (H,)
    const float* ptrW     = (const float*)d_in[19];  // (2H+E+H,)
    const float* ptrb     = (const float*)d_in[20];
    const float* out_W    = (const float*)d_in[21];  // (V, 2H) row-major == colB

    // workspace layout (floats), ~32.4 MB total
    float* ws        = (float*)d_ws;
    float* Xcat      = ws;                    // 128*1024
    float* dec_emb   = Xcat + 131072;         // 128*512
    float* gates     = dec_emb + 65536;       // 128*2048
    float* hidden    = gates + 262144;        // 128*1024
    float* attn_WT   = hidden + 131072;       // 512*1024
    float* attn_W2T  = attn_WT + 524288;      // 512*512
    float* hidW      = attn_W2T + 262144;     // 128*512
    float* attention = hidW + 65536;          // 128*400
    float* enc_ctx   = attention + 51200;     // 128*512
    float* combined  = enc_ctx + 65536;       // 128*1024
    float* logits    = combined + 131072;     // 128*50000

    // d_out: [out_probs | h_new | c_new | enc_attn | prob_ptr]
    float* out      = (float*)d_out;
    float* o_probs  = out;                    // 128*50100
    float* o_h      = o_probs + (size_t)BB * EXTV;
    float* o_c      = o_h + (size_t)BB * HB;
    float* o_attn   = o_c + (size_t)BB * HB;
    float* o_ptr    = o_attn + (size_t)BB * SS;

    // 1. Xcat = [embedded | prev_enc_context]
    concat_kernel<<<(BB * (EB + HB) + 255) / 256, 256, 0, stream>>>(embedded, prevctx, Xcat, EB, HB, BB);
    // 2-3. transposes for attn weights (column-major B for WMMA)
    transpose_kernel<<<(2 * HB * HB + 255) / 256, 256, 0, stream>>>(attn_W, attn_WT, 2 * HB, HB);
    transpose_kernel<<<(HB * HB + 255) / 256, 256, 0, stream>>>(attn_W2, attn_W2T, HB, HB);
    // 4. dec_input_emb = Xcat @ fc_dec_W.T + b
    gemm_bf16_wmma<<<dim3(EB / 16, BB / 16), 32, 0, stream>>>(Xcat, EB + HB, fc_dec_W, EB + HB,
                                                              dec_emb, EB, EB + HB, fc_dec_b, 0);
    // 5-6. gates = dec_emb @ W_ih.T + b_ih + h0 @ W_hh.T + b_hh
    gemm_bf16_wmma<<<dim3(4 * HB / 16, BB / 16), 32, 0, stream>>>(dec_emb, EB, W_ih, EB,
                                                                  gates, 4 * HB, EB, b_ih, 0);
    gemm_bf16_wmma<<<dim3(4 * HB / 16, BB / 16), 32, 0, stream>>>(h0, HB, W_hh, HB,
                                                                  gates, 4 * HB, HB, b_hh, 1);
    // 7. LSTM pointwise; hidden = [h|c]; h,c straight into d_out
    lstm_kernel<<<BB, HB, 0, stream>>>(gates, c0, o_h, o_c, hidden);
    // 8. hidW = hidden @ attn_W
    gemm_bf16_wmma<<<dim3(HB / 16, BB / 16), 32, 0, stream>>>(hidden, 2 * HB, attn_WT, 2 * HB,
                                                              hidW, HB, 2 * HB, nullptr, 0);
    // 9. fused energy/tanh/W3 reduction -> attention scores
    attn_energy_kernel<<<(BB * SS) / 16, 256, 0, stream>>>(enc, attn_W2T, hidW, cov, coverW,
                                                           attn_W3, attention);
    // 10. masked softmax + context; enc_attn into d_out; combined = [h|ctx]
    softmax_ctx_kernel<<<BB, 512, 0, stream>>>(attention, mask, enc, o_h, o_attn, enc_ctx, combined);
    // 11. pointer probability into d_out
    ptr_kernel<<<BB, 256, 0, stream>>>(o_h, enc_ctx, embedded, ptrW, ptrb, o_ptr);
    // 12. logits = combined @ out_W.T — one wave covers all of M so out_W streams HBM once
    gemm_m128_wmma<<<dim3(VV / 16), 32, 0, stream>>>(combined, 2 * HB, out_W, 2 * HB,
                                                     logits, VV, 2 * HB);
    // 13. (1-p_ptr)*softmax(logits) into out_probs, zero ext tail
    vocab_kernel<<<BB, 512, 0, stream>>>(logits, o_ptr, o_probs);
    // 14. scatter-add pointer mass
    scatter_kernel<<<(BB * SS + 255) / 256, 256, 0, stream>>>(widx, o_attn, o_ptr, o_probs);
    // 15. log(out_probs + 1e-31)
    log_kernel<<<((size_t)BB * EXTV + 255) / 256, 256, 0, stream>>>(o_probs, BB * EXTV);
}
